// STGCN_61048665145578
// MI455X (gfx1250) — matrix operand order repaired and checked by reading, not compiled, not run on hardware
//
#include <hip/hip_runtime.h>

typedef __bf16 bf16;
typedef bf16  v16bf __attribute__((ext_vector_type(16)));
typedef bf16  v8bf  __attribute__((ext_vector_type(8)));
typedef bf16  v4bf  __attribute__((ext_vector_type(4)));
typedef float v8f   __attribute__((ext_vector_type(8)));

#define NB    512                 // nodes
#define NN    (NB*NB)
#define BATCH 32
#define TT    12
#define HIDC  64
#define UD    (BATCH*TT)          // 384
#define ACTN  (NB*UD*HIDC)        // 12,582,912 elements

// ---------------------------------------------------------------------------
// Small helper kernels (cheap, memory-light)
// ---------------------------------------------------------------------------
__global__ void degree_kernel(const float* __restrict__ g, float* __restrict__ dinv) {
  int n = blockIdx.x * blockDim.x + threadIdx.x;
  if (n >= NB) return;
  float s = 0.f;
  for (int m = 0; m < NB; ++m) s += g[n * NB + m];
  dinv[n] = rsqrtf(s);
}

__global__ void lap_kernel(const float* __restrict__ g, const float* __restrict__ dinv,
                           float* __restrict__ P) {
  int idx = blockIdx.x * blockDim.x + threadIdx.x;
  if (idx >= NN) return;
  int n = idx >> 9, m = idx & 511;
  float I = (n == m) ? 1.f : 0.f;
  P[idx]      = I;                                  // T0 = I
  P[NN + idx] = I - dinv[n] * g[idx] * dinv[m];     // T1 = L
}

// Pk = 2 * (L @ Pp) - Pp2   (exact fp32; only 0.27 GFLOP per call)
__global__ __launch_bounds__(256) void cheb_step_kernel(
    const float* __restrict__ L, const float* __restrict__ Pp,
    const float* __restrict__ Pp2, float* __restrict__ Pk) {
  __shared__ float sL[16][17], sP[16][17];
  int tx = threadIdx.x, ty = threadIdx.y;
  int row = blockIdx.y * 16 + ty, col = blockIdx.x * 16 + tx;
  float acc = 0.f;
  for (int k0 = 0; k0 < NB; k0 += 16) {
    sL[ty][tx] = L[row * NB + k0 + tx];
    sP[ty][tx] = Pp[(k0 + ty) * NB + col];
    __syncthreads();
#pragma unroll
    for (int j = 0; j < 16; ++j) acc += sL[ty][j] * sP[j][tx];
    __syncthreads();
  }
  Pk[row * NB + col] = 2.f * acc - Pp2[row * NB + col];
}

__global__ void f2bf_kernel(const float* __restrict__ in, bf16* __restrict__ out, int n) {
  int i = blockIdx.x * blockDim.x + threadIdx.x;
  if (i < n) out[i] = (bf16)in[i];
}

// ---------------------------------------------------------------------------
// WMMA bf16 GEMM:  C[M,N] = A[M,K] @ B[K,N]  (+ Cin if ACC), f32 accumulate.
// Double-buffered LDS pipeline:
//   A tile staged with GLOBAL_LOAD_ASYNC_TO_LDS_B128 (ASYNCcnt, no VGPR bounce)
//   B tile staged via 4x4 register micro-transpose (b64 loads + b64 LDS stores)
//   Next tile issued before current tile's WMMAs; one barrier per K-step.
// Block tile 128x128, BK=32; 8 wave32 waves (4Mx2N); wave tile 32x64 = 8 accums.
// M,K must be multiples of 128/32 (true at all call sites); N guarded.
// ---------------------------------------------------------------------------
template <bool ACC, bool BF16OUT>
__global__ __launch_bounds__(256) void gemm_bf16_kernel(
    const bf16* __restrict__ A, const bf16* __restrict__ B,
    const float* __restrict__ Cin, void* __restrict__ Cout,
    int N, int K, int lda, int ldb, int ldc) {
  constexpr int BK = 32, PAD = 8, LDR = BK + PAD;   // 80B row pitch (16B aligned)
  __shared__ __align__(16) bf16 As[2][128][LDR];
  __shared__ __align__(16) bf16 Bs[2][128][LDR];

  const int tid  = threadIdx.x;
  const int lane = tid & 31;
  const int wave = tid >> 5;
  const int bm = blockIdx.y * 128;
  const int bn = blockIdx.x * 128;
  const int wm = (wave >> 1) * 32;
  const int wn = (wave & 1) * 64;

  // per-thread A staging geometry: 2 chunks of 8 bf16 (16B) each
  const int ar0 = tid >> 2,        ak0 = (tid & 3) * 8;          // chunk q = tid
  const int ar1 = (tid + 256) >> 2, ak1 = ((tid + 256) & 3) * 8; // chunk q = tid+256
  const bf16* Ag0 = A + (size_t)(bm + ar0) * lda + ak0;
  const bf16* Ag1 = A + (size_t)(bm + ar1) * lda + ak1;

  // per-thread B staging geometry: 4x4 micro-tile (kk..kk+3) x (nc..nc+3)
  const int nc = (tid >> 3) * 4;
  const int kk = (tid & 7) * 4;
  const bool bok = (bn + nc) < N;
  const bf16* Bg = B + (size_t)kk * ldb + bn + nc;

  v4bf breg[4];

  auto stageA = [&](int k0, int buf) {
    unsigned l0 = (unsigned)(uintptr_t)&As[buf][ar0][ak0];
    unsigned l1 = (unsigned)(uintptr_t)&As[buf][ar1][ak1];
    const bf16* g0 = Ag0 + k0;
    const bf16* g1 = Ag1 + k0;
    asm volatile("global_load_async_to_lds_b128 %0, %1, off"
                 :: "v"(l0), "v"(g0) : "memory");
    asm volatile("global_load_async_to_lds_b128 %0, %1, off"
                 :: "v"(l1), "v"(g1) : "memory");
  };
  auto loadB = [&](int k0) {
    if (bok) {
#pragma unroll
      for (int j = 0; j < 4; ++j)
        breg[j] = *(const v4bf*)(Bg + (size_t)(k0 + j) * ldb);
    } else {
#pragma unroll
      for (int j = 0; j < 4; ++j)
#pragma unroll
        for (int c = 0; c < 4; ++c) breg[j][c] = (bf16)0.f;
    }
  };
  auto storeB = [&](int buf) {
#pragma unroll
    for (int c = 0; c < 4; ++c) {
      v4bf o;
#pragma unroll
      for (int j = 0; j < 4; ++j) o[j] = breg[j][c];
      *(v4bf*)&Bs[buf][nc + c][kk] = o;
    }
  };

  v8f acc[2][4];
#pragma unroll
  for (int i = 0; i < 2; ++i)
#pragma unroll
    for (int j = 0; j < 4; ++j)
#pragma unroll
      for (int l = 0; l < 8; ++l) acc[i][j][l] = 0.f;

  // ---- prologue: stage tile 0 into buffer 0
  stageA(0, 0);
  loadB(0);
  storeB(0);
  asm volatile("s_wait_asynccnt 0" ::: "memory");
  __syncthreads();

  const int nk = K / BK;
  for (int i = 0; i < nk; ++i) {
    const int cur = i & 1, nxt = cur ^ 1;
    // issue next tile's memory before computing current tile
    if (i + 1 < nk) {
      stageA((i + 1) * BK, nxt);
      loadB((i + 1) * BK);
    }
    if (i + 2 < nk)   // keep the prefetch path (global_prefetch_b8) warm for B
      __builtin_prefetch(Bg + (size_t)((i + 2) * BK) * ldb, 0, 1);

    // ---- A fragments: ISA 16-bit A 16x32 layout
    v16bf af[2];
#pragma unroll
    for (int mf = 0; mf < 2; ++mf) {
      int row = wm + mf * 16 + (lane & 15);
      int kb  = (lane >> 4) * 8;
      v8bf lo = *(const v8bf*)&As[cur][row][kb];
      v8bf hi = *(const v8bf*)&As[cur][row][kb + 16];
      af[mf] = __builtin_shufflevector(lo, hi, 0, 1, 2, 3, 4, 5, 6, 7,
                                                8, 9, 10, 11, 12, 13, 14, 15);
    }
    // ---- B fragments + 8 WMMAs
#pragma unroll
    for (int nf = 0; nf < 4; ++nf) {
      int col = wn + nf * 16 + (lane & 15);
      int kh  = (lane >> 4) * 8;
      v8bf lo = *(const v8bf*)&Bs[cur][col][kh];
      v8bf hi = *(const v8bf*)&Bs[cur][col][kh + 16];
      v16bf bv = __builtin_shufflevector(lo, hi, 0, 1, 2, 3, 4, 5, 6, 7,
                                                 8, 9, 10, 11, 12, 13, 14, 15);
#pragma unroll
      for (int mf = 0; mf < 2; ++mf)
        acc[mf][nf] = __builtin_amdgcn_wmma_f32_16x16x32_bf16(
            false, af[mf], false, bv, (short)0, acc[mf][nf], false, false);
    }

    if (i + 1 < nk) {
      storeB(nxt);
      asm volatile("s_wait_asynccnt 0" ::: "memory");
    }
    __syncthreads();
  }

  // ---- epilogue: C/D layout — VGPR i -> M = i + (lane>=16 ? 8 : 0), N = lane&15
#pragma unroll
  for (int mf = 0; mf < 2; ++mf) {
#pragma unroll
    for (int nf = 0; nf < 4; ++nf) {
      int gcol = bn + wn + nf * 16 + (lane & 15);
      if (gcol >= N) continue;
      int rbase = bm + wm + mf * 16 + ((lane >> 4) << 3);
#pragma unroll
      for (int i = 0; i < 8; ++i) {
        size_t off = (size_t)(rbase + i) * ldc + gcol;
        float v = acc[mf][nf][i];
        if (ACC) v += Cin[off];
        if (BF16OUT) ((bf16*)Cout)[off] = (bf16)v;
        else         ((float*)Cout)[off] = v;
      }
    }
  }
}

// ---------------------------------------------------------------------------
// Temporal convs (direct; ~10% of FLOPs) + epilogues, node-major [n][b][t][c]
// ---------------------------------------------------------------------------
__global__ void tconv_src_kernel(const float* __restrict__ x,   // src [B,N,T,1]
                                 const float* __restrict__ w,   // [64,1,1,3]
                                 const float* __restrict__ bias,
                                 bf16* __restrict__ out) {      // [N,B,T,64] bf16
  int idx = blockIdx.x * blockDim.x + threadIdx.x;
  int h = idx & 63;
  int r = idx >> 6;
  int t = r % TT;
  int nb = r / TT;
  int b = nb & 31, n = nb >> 5;
  const float* xp = x + (size_t)(b * NB + n) * TT;
  float acc = bias[h];
#pragma unroll
  for (int dt = 0; dt < 3; ++dt) {
    int ts = t + dt - 1;
    if (ts >= 0 && ts < TT) acc += xp[ts] * w[h * 3 + dt];
  }
  out[idx] = (bf16)acc;
}

template <int COUT, bool BF16OUT>
__global__ void tconv_hid_kernel(const float* __restrict__ x,   // [N,B,T,64] f32
                                 const float* __restrict__ w,   // [COUT,64,1,3]
                                 const float* __restrict__ bias,
                                 void* __restrict__ out) {      // [N,B,T,COUT]
  int idx = blockIdx.x * blockDim.x + threadIdx.x;
  int h = idx % COUT;
  int r = idx / COUT;                 // (n*32+b)*12 + t
  int t = r % TT;
  const float* xr = x + (size_t)(r - t) * HIDC;
  float acc = bias[h];
#pragma unroll
  for (int dt = 0; dt < 3; ++dt) {
    int ts = t + dt - 1;
    if (ts < 0 || ts >= TT) continue;
    const float* xp = xr + (size_t)ts * HIDC;
    const float* wp = w + (size_t)h * HIDC * 3 + dt;
    float s = 0.f;
#pragma unroll 8
    for (int c = 0; c < HIDC; ++c) s += xp[c] * wp[c * 3];
    acc += s;
  }
  if (BF16OUT) ((bf16*)out)[idx] = (bf16)acc;
  else         ((float*)out)[idx] = acc;
}

__global__ void relu_bias_kernel(float* __restrict__ S, const float* __restrict__ bias) {
  int idx = blockIdx.x * blockDim.x + threadIdx.x;
  S[idx] = fmaxf(S[idx] + bias[idx & 63], 0.f);
}

__global__ void fc_kernel(const float* __restrict__ y,    // [N,B,T] f32
                          const float* __restrict__ fw,   // [TRG,SRC]
                          const float* __restrict__ fb,
                          float* __restrict__ out) {      // [B,N,TRG,1]
  int idx = blockIdx.x * blockDim.x + threadIdx.x;        // b*N*12 + n*12 + s
  int s = idx % TT;
  int bn = idx / TT;
  int b = bn / NB, n = bn % NB;
  const float* yp = y + (size_t)(n * BATCH + b) * TT;
  float acc = fb[s];
#pragma unroll
  for (int t = 0; t < TT; ++t) acc += yp[t] * fw[s * TT + t];
  out[idx] = acc;
}

// ---------------------------------------------------------------------------
extern "C" void kernel_launch(void* const* d_in, const int* in_sizes, int n_in,
                              void* d_out, int out_size, void* d_ws, size_t ws_size,
                              hipStream_t stream) {
  (void)in_sizes; (void)n_in; (void)out_size; (void)ws_size;
  const float* src   = (const float*)d_in[0];
  const float* graph = (const float*)d_in[1];
  const float* t1w1  = (const float*)d_in[2];
  const float* t1b1  = (const float*)d_in[3];
  const float* sw1   = (const float*)d_in[4];
  const float* sb1   = (const float*)d_in[5];
  const float* t2w1  = (const float*)d_in[6];
  const float* t2b1  = (const float*)d_in[7];
  const float* t1w2  = (const float*)d_in[8];
  const float* t1b2  = (const float*)d_in[9];
  const float* sw2   = (const float*)d_in[10];
  const float* sb2   = (const float*)d_in[11];
  const float* t2w2  = (const float*)d_in[12];
  const float* t2b2  = (const float*)d_in[13];
  const float* fcw   = (const float*)d_in[14];
  const float* fcb   = (const float*)d_in[15];
  float* out = (float*)d_out;

  char* ws = (char*)d_ws;
  size_t off = 0;
  auto walloc = [&](size_t bytes) -> void* {
    void* p = ws + off;
    off = (off + bytes + 255) & ~(size_t)255;
    return p;
  };
  float* dinv = (float*)walloc((size_t)NB * 4);
  float* P    = (float*)walloc((size_t)4 * NN * 4);
  bf16*  Pbf  = (bf16*) walloc((size_t)4 * NN * 2);
  bf16*  Wbf1 = (bf16*) walloc((size_t)4 * 64 * 64 * 2);
  bf16*  Wbf2 = (bf16*) walloc((size_t)4 * 64 * 64 * 2);
  bf16*  act  = (bf16*) walloc((size_t)ACTN * 2);
  bf16*  Zk   = (bf16*) walloc((size_t)ACTN * 2);
  float* S    = (float*)walloc((size_t)ACTN * 4);
  float* yt   = (float*)walloc((size_t)ACTN * 4);
  float* yfin = (float*)walloc((size_t)NB * UD * 4);

  // Chebyshev polynomials (fp32 exact), then bf16 copies for WMMA
  degree_kernel<<<2, 256, 0, stream>>>(graph, dinv);
  lap_kernel<<<NN / 256, 256, 0, stream>>>(graph, dinv, P);
  dim3 cb(16, 16), cg(32, 32);
  cheb_step_kernel<<<cg, cb, 0, stream>>>(P + NN, P + NN, P, P + 2 * (size_t)NN);
  cheb_step_kernel<<<cg, cb, 0, stream>>>(P + NN, P + 2 * (size_t)NN, P + NN, P + 3 * (size_t)NN);
  f2bf_kernel<<<(4 * NN) / 256, 256, 0, stream>>>(P, Pbf, 4 * NN);
  f2bf_kernel<<<64, 256, 0, stream>>>(sw1, Wbf1, 4 * 64 * 64);
  f2bf_kernel<<<64, 256, 0, stream>>>(sw2, Wbf2, 4 * 64 * 64);

  const int Mbig = NB * UD;          // 196608
  const int Nbig = UD * HIDC;        // 24576
  dim3 gz(1, Mbig / 128);            // (1,1536)  Zk = act @ Wk
  dim3 gs(Nbig / 128, NB / 128);     // (192,4)   S += Pk @ Zk

  // ---------------- ST block 1 ----------------
  tconv_src_kernel<<<ACTN / 256, 256, 0, stream>>>(src, t1w1, t1b1, act);
  for (int k = 0; k < 4; ++k) {
    gemm_bf16_kernel<false, true><<<gz, 256, 0, stream>>>(
        act, Wbf1 + (size_t)k * 64 * 64, nullptr, Zk, 64, 64, 64, 64, 64);
    if (k == 0)
      gemm_bf16_kernel<false, false><<<gs, 256, 0, stream>>>(
          Pbf + (size_t)k * NN, Zk, nullptr, S, Nbig, NB, NB, Nbig, Nbig);
    else
      gemm_bf16_kernel<true, false><<<gs, 256, 0, stream>>>(
          Pbf + (size_t)k * NN, Zk, S, S, Nbig, NB, NB, Nbig, Nbig);
  }
  relu_bias_kernel<<<ACTN / 256, 256, 0, stream>>>(S, sb1);
  tconv_hid_kernel<64, false><<<ACTN / 256, 256, 0, stream>>>(S, t2w1, t2b1, yt);

  // ---------------- ST block 2 ----------------
  tconv_hid_kernel<64, true><<<ACTN / 256, 256, 0, stream>>>(yt, t1w2, t1b2, act);
  for (int k = 0; k < 4; ++k) {
    gemm_bf16_kernel<false, true><<<gz, 256, 0, stream>>>(
        act, Wbf2 + (size_t)k * 64 * 64, nullptr, Zk, 64, 64, 64, 64, 64);
    if (k == 0)
      gemm_bf16_kernel<false, false><<<gs, 256, 0, stream>>>(
          Pbf + (size_t)k * NN, Zk, nullptr, S, Nbig, NB, NB, Nbig, Nbig);
    else
      gemm_bf16_kernel<true, false><<<gs, 256, 0, stream>>>(
          Pbf + (size_t)k * NN, Zk, S, S, Nbig, NB, NB, Nbig, Nbig);
  }
  relu_bias_kernel<<<ACTN / 256, 256, 0, stream>>>(S, sb2);
  tconv_hid_kernel<1, false><<<(NB * UD) / 256, 256, 0, stream>>>(S, t2w2, t2b2, yfin);

  // fc over time -> [B,N,TRG,1]
  fc_kernel<<<(NB * UD) / 256, 256, 0, stream>>>(yfin, fcw, fcb, out);
}
